// TransformerLayerN2N_90623809946411
// MI455X (gfx1250) — matrix operand order repaired and checked by reading, not compiled, not run on hardware
//
#include <hip/hip_runtime.h>
#include <math.h>

// ---------------- problem constants ----------------
#define B_   2
#define NV_  6
#define M_   512
#define NS_  512
#define H_   4
#define HD_  64
#define D_   256
#define R_   (B_*NV_*M_)      // 6144 tokens per tensor

typedef __bf16 bf16;
typedef __attribute__((ext_vector_type(16))) __bf16 v16bf;
typedef __attribute__((ext_vector_type(8)))  float  v8f;

union FragB { v16bf v; uint4 u[2]; };

static __device__ inline v8f wmma_bf16(v16bf a, v16bf b, v8f c) {
  // v_wmma_f32_16x16x32_bf16 : D = A(16x32) * B(32x16) + C
  return __builtin_amdgcn_wmma_f32_16x16x32_bf16(false, a, false, b, (short)0, c, false, false);
}

// CDNA5 async global->LDS copy (ASYNCcnt-tracked). lds_off = low 32 bits of the
// generic shared-memory address (hardware truncates addr[31:0] for LDS).
static __device__ inline void async_g2l_b128(unsigned lds_off, const void* gptr) {
  asm volatile("global_load_async_to_lds_b128 %0, %1, off"
               :: "v"(lds_off), "v"(gptr) : "memory");
}
static __device__ inline void wait_async0() {
  asm volatile("s_wait_asynccnt 0x0" ::: "memory");
}
static __device__ inline unsigned lds_off32(const void* p) {
  return (unsigned)(uintptr_t)p;
}

// ---------------- elementwise helpers ----------------
__global__ void cvt_f32_bf16_kernel(const float* __restrict__ src, bf16* __restrict__ dst, long long n) {
  long long i = (long long)blockIdx.x * blockDim.x + threadIdx.x;
  if (i < n) dst[i] = (bf16)src[i];
}

// weight convert + transpose: src[K][N] f32 -> dst[N][K] bf16
__global__ void cvt_transpose_bf16_kernel(const float* __restrict__ src, bf16* __restrict__ dst,
                                          int K, int N) {
  long long i = (long long)blockIdx.x * blockDim.x + threadIdx.x;
  if (i >= (long long)K * N) return;
  int k = (int)(i / N), n = (int)(i % N);
  dst[(size_t)n * K + k] = (bf16)src[i];
}

__global__ void fill_bf16_kernel(bf16* __restrict__ dst, long long n) {
  long long i = (long long)blockIdx.x * blockDim.x + threadIdx.x;
  if (i < n) dst[i] = (bf16)0.0f;
}

__global__ void add_res_kernel(const float* __restrict__ a, const float* __restrict__ b,
                               float* __restrict__ o, long long n) {
  long long i = (long long)blockIdx.x * blockDim.x + threadIdx.x;
  if (i < n) o[i] = a[i] + b[i];
}

__global__ void concat_bf16_kernel(const float* __restrict__ x, const float* __restrict__ msg,
                                   bf16* __restrict__ o, long long n /* R*512 */) {
  long long i = (long long)blockIdx.x * blockDim.x + threadIdx.x;
  if (i >= n) return;
  long long row = i >> 9;
  int col = (int)(i & 511);
  float v = (col < D_) ? x[row * D_ + col] : msg[row * D_ + (col - D_)];
  o[i] = (bf16)v;
}

// ---------------- GEMM: C[M,N] = A[M,K](bf16) * Wt^T + bias ----------------
// Wt is the PRE-TRANSPOSED weight: Wt[N][K] bf16 (so B tiles are row-contiguous).
// block = 256 threads (8 waves). Tile 128x64, K step 32. M%128==0, N%64==0, K%32==0.
// A/B tiles staged global->LDS with GLOBAL_LOAD_ASYNC_TO_LDS_B128 (ASYNCcnt).
__global__ __launch_bounds__(256)
void gemm_bf16_kernel(const bf16* __restrict__ A, const bf16* __restrict__ Wt,
                      const float* __restrict__ bias, float* __restrict__ C,
                      int M, int N, int K) {
  __shared__ bf16 Al[128][40];   // padded rows (80B) to dodge bank conflicts
  __shared__ bf16 Bl[64][40];    // Bl[n][k]
  const int t    = threadIdx.x;
  const int lane = t & 31;
  const int wave = t >> 5;       // 0..7
  const int wm   = wave >> 1;    // 0..3 -> rows 32*wm
  const int wn   = wave & 1;     // 0..1 -> cols 32*wn
  const int m0   = blockIdx.y * 128;
  const int n0   = blockIdx.x * 64;
  const int h    = lane >> 4;
  const int l16  = lane & 15;

  // per-thread staging addresses (constant across K loop except k0 advance)
  const int arow = t >> 1, ahalf = t & 1;
  const unsigned aLds = lds_off32(&Al[arow][16 * ahalf]);
  const int brow = t >> 2, bq = t & 3;
  const unsigned bLds = lds_off32(&Bl[brow][8 * bq]);

  v8f acc[2][2];
#pragma unroll
  for (int a = 0; a < 2; ++a)
#pragma unroll
    for (int b = 0; b < 2; ++b)
#pragma unroll
      for (int r = 0; r < 8; ++r) acc[a][b][r] = 0.0f;

  for (int k0 = 0; k0 < K; k0 += 32) {
    { // A tile 128x32: 2 x 16B async per thread
      const bf16* g = A + (size_t)(m0 + arow) * K + k0 + 16 * ahalf;
      async_g2l_b128(aLds, g);
      async_g2l_b128(aLds + 16u, g + 8);
    }
    { // B tile 64x32 from transposed weights: 1 x 16B async per thread
      const bf16* g = Wt + (size_t)(n0 + brow) * K + k0 + 8 * bq;
      async_g2l_b128(bLds, g);
    }
    wait_async0();
    __syncthreads();

    FragB af[2], bfr[2];
#pragma unroll
    for (int mt = 0; mt < 2; ++mt) {  // A frag: lane holds contiguous K = 16h..16h+15
      const bf16* p = &Al[32 * wm + 16 * mt + l16][0];
      af[mt].u[0] = *(const uint4*)(p + 16 * h);
      af[mt].u[1] = *(const uint4*)(p + 16 * h + 8);
    }
#pragma unroll
    for (int nt = 0; nt < 2; ++nt) {  // B frag: lane holds contiguous K = 16h..16h+15
      const bf16* p = &Bl[32 * wn + 16 * nt + l16][0];
      bfr[nt].u[0] = *(const uint4*)(p + 16 * h);
      bfr[nt].u[1] = *(const uint4*)(p + 16 * h + 8);
    }
#pragma unroll
    for (int mt = 0; mt < 2; ++mt)
#pragma unroll
      for (int nt = 0; nt < 2; ++nt)
        acc[mt][nt] = wmma_bf16(af[mt].v, bfr[nt].v, acc[mt][nt]);
    __syncthreads();
  }

#pragma unroll
  for (int mt = 0; mt < 2; ++mt)
#pragma unroll
    for (int nt = 0; nt < 2; ++nt) {
      int col = n0 + 32 * wn + 16 * nt + l16;
      float bb = bias[col];
      int rb = m0 + 32 * wm + 16 * mt + 8 * h;  // C layout: row = r + 8*(lane>>4)
#pragma unroll
      for (int r = 0; r < 8; ++r)
        C[(size_t)(rb + r) * N + col] = acc[mt][nt][r] + bb;
    }
}

// ---------------- fused flash attention (bf16 in, bf16 out, f32 softmax/acc) --------
// q:[batch][Lq][64] k:[batch][Lk][64] vt:[batch][64][LkPad] (V transposed, pads zeroed)
// out index = i0*s0 + i1*s1 + i2*s2 + row*sRow + d   with batch=(i0*nMid+i1)*nInner+i2
__global__ __launch_bounds__(128)
void attn_kernel(const bf16* __restrict__ q, const bf16* __restrict__ k,
                 const bf16* __restrict__ vt, bf16* __restrict__ out,
                 int nBatch, int Lq, int Lk, int LkPad, float scale,
                 int nMid, int nInner,
                 long long s0, long long s1, long long s2, long long sRow) {
  __shared__ bf16 Pl[4][16][40];  // wave-private P transpose buffer
  const int wave = threadIdx.x >> 5, lane = threadIdx.x & 31;
  const int h = lane >> 4, l16 = lane & 15;
  const int qTiles = (Lq + 15) >> 4;
  long long task = (long long)blockIdx.x * 4 + wave;
  if (task >= (long long)nBatch * qTiles) return;   // whole-wave uniform exit
  const int batch = (int)(task / qTiles);
  const int m0 = (int)(task % qTiles) << 4;

  const bf16* qB = q  + (size_t)batch * Lq * 64;
  const bf16* kB = k  + (size_t)batch * Lk * 64;
  const bf16* vB = vt + (size_t)batch * 64 * LkPad;

  FragB qf[2];
  {
    int mr = m0 + l16; if (mr >= Lq) mr = Lq - 1;
    const bf16* p = qB + (size_t)mr * 64;
    qf[0].u[0] = *(const uint4*)(p + 16 * h);       qf[0].u[1] = *(const uint4*)(p + 16 * h + 8);
    qf[1].u[0] = *(const uint4*)(p + 32 + 16 * h);  qf[1].u[1] = *(const uint4*)(p + 40 + 16 * h);
  }

  v8f o[4];
  float rowM[8], rowS[8];
#pragma unroll
  for (int dt = 0; dt < 4; ++dt)
#pragma unroll
    for (int r = 0; r < 8; ++r) o[dt][r] = 0.0f;
#pragma unroll
  for (int r = 0; r < 8; ++r) { rowM[r] = -3.0e38f; rowS[r] = 0.0f; }

  for (int j0 = 0; j0 < Lk; j0 += 32) {
    v8f s[2];
#pragma unroll
    for (int th = 0; th < 2; ++th) {   // S = Q * K^T ; B-frag lane = key column
      int kr = j0 + 16 * th + l16; if (kr >= Lk) kr = Lk - 1;
      const bf16* p = kB + (size_t)kr * 64;
      FragB kf0, kf1;
      kf0.u[0] = *(const uint4*)(p + 16 * h);       kf0.u[1] = *(const uint4*)(p + 16 * h + 8);
      kf1.u[0] = *(const uint4*)(p + 32 + 16 * h);  kf1.u[1] = *(const uint4*)(p + 40 + 16 * h);
      v8f zz;
#pragma unroll
      for (int r = 0; r < 8; ++r) zz[r] = 0.0f;
      v8f sv = wmma_bf16(qf[0].v, kf0.v, zz);
      sv = wmma_bf16(qf[1].v, kf1.v, sv);
      bool colValid = (j0 + 16 * th + l16) < Lk;
#pragma unroll
      for (int r = 0; r < 8; ++r) sv[r] = colValid ? sv[r] * scale : -1.0e30f;
      s[th] = sv;
    }

    float p0[8], p1[8], alpha[8];
#pragma unroll
    for (int r = 0; r < 8; ++r) {     // per-row online softmax (rows live per lane-half)
      float bm = fmaxf(s[0][r], s[1][r]);
      bm = fmaxf(bm, __shfl_xor(bm, 1)); bm = fmaxf(bm, __shfl_xor(bm, 2));
      bm = fmaxf(bm, __shfl_xor(bm, 4)); bm = fmaxf(bm, __shfl_xor(bm, 8));
      float nm = fmaxf(rowM[r], bm);
      alpha[r] = __expf(rowM[r] - nm);
      rowM[r] = nm;
      p0[r] = __expf(s[0][r] - nm);
      p1[r] = __expf(s[1][r] - nm);
      float bs = p0[r] + p1[r];
      bs += __shfl_xor(bs, 1); bs += __shfl_xor(bs, 2);
      bs += __shfl_xor(bs, 4); bs += __shfl_xor(bs, 8);
      rowS[r] = rowS[r] * alpha[r] + bs;
    }
#pragma unroll
    for (int dt = 0; dt < 4; ++dt)
#pragma unroll
      for (int r = 0; r < 8; ++r) o[dt][r] *= alpha[r];

    // P (C layout) -> LDS -> A layout
#pragma unroll
    for (int r = 0; r < 8; ++r) {
      Pl[wave][r + 8 * h][l16]      = (bf16)p0[r];
      Pl[wave][r + 8 * h][16 + l16] = (bf16)p1[r];
    }
    asm volatile("s_wait_dscnt 0" ::: "memory");
    FragB pf;
    {
      const bf16* p = &Pl[wave][l16][0];
      pf.u[0] = *(const uint4*)(p + 16 * h);
      pf.u[1] = *(const uint4*)(p + 16 * h + 8);
    }
#pragma unroll
    for (int dt = 0; dt < 4; ++dt) {  // O += P * V ; V pre-transposed, contiguous runs
      const bf16* p = vB + (size_t)(16 * dt + l16) * LkPad + j0;
      FragB vf;
      vf.u[0] = *(const uint4*)(p + 16 * h);
      vf.u[1] = *(const uint4*)(p + 16 * h + 8);
      o[dt] = wmma_bf16(pf.v, vf.v, o[dt]);
    }
  }

  const int i2 = batch % nInner;
  const int rem = batch / nInner;
  const int i1 = rem % nMid;
  const int i0 = rem / nMid;
  const long long base = i0 * s0 + i1 * s1 + i2 * s2;
#pragma unroll
  for (int r = 0; r < 8; ++r) {
    int row = m0 + r + 8 * h;
    if (row >= Lq) continue;
    float inv = 1.0f / rowS[r];
#pragma unroll
    for (int dt = 0; dt < 4; ++dt)
      out[base + (long long)row * sRow + 16 * dt + l16] = (bf16)(o[dt][r] * inv);
  }
}

// ---------------- gathers (+RoPE) ----------------
// qkv rows: [..., H, HD, 3]; rope: y[2i]=x0*f0[2i]-x1*f1[2i]; y[2i+1]=x1*f0[2i+1]+x0*f1[2i+1]
__global__ void gather_rope_self_kernel(const float* __restrict__ qkv, const float* __restrict__ enc,
                                        bf16* __restrict__ qb, bf16* __restrict__ kb,
                                        bf16* __restrict__ vt, int Mr) {
  long long idx = (long long)blockIdx.x * blockDim.x + threadIdx.x;
  long long total = (long long)B_ * NV_ * H_ * Mr * 32;
  if (idx >= total) return;
  int dp = (int)(idx % 32); long long r = idx / 32;
  int m = (int)(r % Mr); r /= Mr;
  int hh = (int)(r % H_); r /= H_;
  int vv = (int)(r % NV_); int b = (int)(r / NV_);
  int d0 = dp * 2;
  size_t qi = ((((size_t)(b * NV_ + vv) * Mr + m) * H_ + hh) * HD_ + d0) * 3;
  float q0 = qkv[qi],     k0 = qkv[qi + 1], v0 = qkv[qi + 2];
  float q1 = qkv[qi + 3], k1 = qkv[qi + 4], v1 = qkv[qi + 5];
  size_t ei = ((size_t)vv * Mr + m) * HD_ + d0;        // freq(view,m,d)
  size_t eo = (size_t)NV_ * Mr * HD_;
  float f00 = enc[ei], f01 = enc[ei + 1], f10 = enc[eo + ei], f11 = enc[eo + ei + 1];
  float rq0 = q0 * f00 - q1 * f10, rq1 = q1 * f01 + q0 * f11;
  float rk0 = k0 * f00 - k1 * f10, rk1 = k1 * f01 + k0 * f11;
  size_t ob = (((size_t)(b * NV_ + vv) * H_ + hh) * Mr + m) * HD_ + d0;
  qb[ob] = (bf16)rq0; qb[ob + 1] = (bf16)rq1;
  kb[ob] = (bf16)rk0; kb[ob + 1] = (bf16)rk1;
  size_t vb = ((size_t)(b * NV_ + vv) * H_ + hh) * ((size_t)HD_ * Mr) + (size_t)d0 * Mr + m;
  vt[vb] = (bf16)v0; vt[vb + Mr] = (bf16)v1;
}

__global__ void gather_t2t_kernel(const float* __restrict__ qkv,
                                  bf16* __restrict__ qb, bf16* __restrict__ kb,
                                  bf16* __restrict__ vt) {
  long long idx = (long long)blockIdx.x * blockDim.x + threadIdx.x;
  long long total = (long long)B_ * M_ * H_ * NV_ * 32;
  if (idx >= total) return;
  int dp = (int)(idx % 32); long long r = idx / 32;
  int vv = (int)(r % NV_); r /= NV_;
  int hh = (int)(r % H_); r /= H_;
  int m = (int)(r % M_); int b = (int)(r / M_);
  int d0 = dp * 2;
  size_t qi = ((((size_t)(b * NV_ + vv) * M_ + m) * H_ + hh) * HD_ + d0) * 3;
  float q0 = qkv[qi],     k0 = qkv[qi + 1], v0 = qkv[qi + 2];
  float q1 = qkv[qi + 3], k1 = qkv[qi + 4], v1 = qkv[qi + 5];
  size_t ob = (((size_t)(b * M_ + m) * H_ + hh) * NV_ + vv) * HD_ + d0;
  qb[ob] = (bf16)q0; qb[ob + 1] = (bf16)q1;
  kb[ob] = (bf16)k0; kb[ob + 1] = (bf16)k1;
  size_t vb = ((size_t)(b * M_ + m) * H_ + hh) * (size_t)(HD_ * 32) + (size_t)d0 * 32 + vv;
  vt[vb] = (bf16)v0; vt[vb + 32] = (bf16)v1;
}

__global__ void gather_s2s_kv_kernel(const float* __restrict__ qkv, const float* __restrict__ kEnc,
                                     bf16* __restrict__ kb, bf16* __restrict__ vt) {
  long long idx = (long long)blockIdx.x * blockDim.x + threadIdx.x;
  long long total = (long long)B_ * H_ * NV_ * NS_ * 32;
  if (idx >= total) return;
  int dp = (int)(idx % 32); long long r = idx / 32;
  int n = (int)(r % NS_); r /= NS_;
  int i = (int)(r % NV_); r /= NV_;
  int hh = (int)(r % H_); int b = (int)(r / H_);
  int d0 = dp * 2;
  size_t qi = ((((size_t)(b * NV_ + i) * NS_ + n) * H_ + hh) * HD_ + d0) * 3;
  float k0 = qkv[qi + 1], k1 = qkv[qi + 4], v0 = qkv[qi + 2], v1 = qkv[qi + 5];
  size_t ei = ((size_t)i * NS_ + n) * HD_ + d0;        // freq(i,n,d)
  size_t eo = (size_t)NV_ * NS_ * HD_;
  float f00 = kEnc[ei], f01 = kEnc[ei + 1], f10 = kEnc[eo + ei], f11 = kEnc[eo + ei + 1];
  float rk0 = k0 * f00 - k1 * f10, rk1 = k1 * f01 + k0 * f11;
  size_t ob = (((size_t)(b * H_ + hh) * NV_ + i) * NS_ + n) * HD_ + d0;
  kb[ob] = (bf16)rk0; kb[ob + 1] = (bf16)rk1;
  size_t vb = ((size_t)(b * H_ + hh) * NV_ + i) * (size_t)(HD_ * NS_) + (size_t)d0 * NS_ + n;
  vt[vb] = (bf16)v0; vt[vb + NS_] = (bf16)v1;
}

__global__ void gather_s2s_q_kernel(const float* __restrict__ qkv, const float* __restrict__ qEnc,
                                    bf16* __restrict__ qb) {
  long long idx = (long long)blockIdx.x * blockDim.x + threadIdx.x;
  long long total = (long long)B_ * H_ * NV_ * NV_ * NS_ * 32;
  if (idx >= total) return;
  int dp = (int)(idx % 32); long long r = idx / 32;
  int n = (int)(r % NS_); r /= NS_;
  int j = (int)(r % NV_); r /= NV_;
  int i = (int)(r % NV_); r /= NV_;
  int hh = (int)(r % H_); int b = (int)(r / H_);
  int d0 = dp * 2;
  size_t qi = ((((size_t)(b * NV_ + j) * NS_ + n) * H_ + hh) * HD_ + d0) * 3;
  float q0 = qkv[qi], q1 = qkv[qi + 3];
  size_t ei = ((size_t)i * NV_ + j) * HD_ + d0;        // freq(i,j,d)
  size_t eo = (size_t)NV_ * NV_ * HD_;
  float f00 = qEnc[ei], f01 = qEnc[ei + 1], f10 = qEnc[eo + ei], f11 = qEnc[eo + ei + 1];
  float rq0 = q0 * f00 - q1 * f10, rq1 = q1 * f01 + q0 * f11;
  size_t ob = (((size_t)(b * H_ + hh) * NV_ + i) * (size_t)(NV_ * NS_) + (size_t)j * NS_ + n) * HD_ + d0;
  qb[ob] = (bf16)rq0; qb[ob + 1] = (bf16)rq1;
}

__global__ void gather_cross_kernel(const float* __restrict__ qF, const float* __restrict__ kvF,
                                    bf16* __restrict__ qb, bf16* __restrict__ kb,
                                    bf16* __restrict__ vt) {
  long long idx = (long long)blockIdx.x * blockDim.x + threadIdx.x;
  long long total = (long long)B_ * NV_ * H_ * M_ * 32;
  if (idx >= total) return;
  int dp = (int)(idx % 32); long long r = idx / 32;
  int m = (int)(r % M_); r /= M_;
  int hh = (int)(r % H_); r /= H_;
  int vv = (int)(r % NV_); int b = (int)(r / NV_);
  int d0 = dp * 2;
  size_t tok = (size_t)(b * NV_ + vv) * M_ + m;
  float q0 = qF[tok * D_ + hh * HD_ + d0], q1 = qF[tok * D_ + hh * HD_ + d0 + 1];
  size_t kvi = tok * (2 * D_) + (size_t)(hh * HD_ + d0) * 2;
  float k0 = kvF[kvi], v0 = kvF[kvi + 1], k1 = kvF[kvi + 2], v1 = kvF[kvi + 3];
  size_t ob = (((size_t)(b * NV_ + vv) * H_ + hh) * M_ + m) * HD_ + d0;
  qb[ob] = (bf16)q0; qb[ob + 1] = (bf16)q1;
  kb[ob] = (bf16)k0; kb[ob + 1] = (bf16)k1;
  size_t vb = ((size_t)(b * NV_ + vv) * H_ + hh) * (size_t)(HD_ * M_) + (size_t)d0 * M_ + m;
  vt[vb] = (bf16)v0; vt[vb + M_] = (bf16)v1;
}

// off-diagonal reshape-mean of s2s context: ctx [b,H,i,(j*NS+n),64] -> out [b,r,n,H*64]
__global__ void s2s_aggregate_kernel(const bf16* __restrict__ ctx, bf16* __restrict__ out) {
  long long idx = (long long)blockIdx.x * blockDim.x + threadIdx.x;
  long long total = (long long)R_ * D_;
  if (idx >= total) return;
  int col = (int)(idx & 255);
  int d = col & 63, hh = col >> 6;
  long long tok = idx >> 8;
  int n = (int)(tok % NS_); long long t2 = tok / NS_;
  int rr = (int)(t2 % NV_); int b = (int)(t2 / NV_);
  float s = 0.0f;
#pragma unroll
  for (int g = 0; g < NV_ - 1; ++g) {
    int f = g * NV_ + rr;
    int i = f / (NV_ - 1);
    int jp = f % (NV_ - 1);
    int j = jp + (jp >= i ? 1 : 0);
    size_t ci = (((size_t)(b * H_ + hh) * NV_ + i) * (size_t)(NV_ * NS_) + (size_t)j * NS_ + n) * HD_ + d;
    s += (float)ctx[ci];
  }
  out[idx] = (bf16)(s * (1.0f / (NV_ - 1)));
}

// ---------------- LayerNorm(512) + exact GELU, wave per row ----------------
__global__ __launch_bounds__(128)
void ln_gelu_kernel(const float* __restrict__ x, const float* __restrict__ g,
                    const float* __restrict__ b, bf16* __restrict__ y, int R) {
  int wave = threadIdx.x >> 5, lane = threadIdx.x & 31;
  int row = blockIdx.x * 4 + wave;
  if (row >= R) return;
  const float* xr = x + (size_t)row * 512;
  float vals[16], s = 0.0f, s2 = 0.0f;
#pragma unroll
  for (int i = 0; i < 16; ++i) {
    float v = xr[lane + 32 * i];
    vals[i] = v; s += v; s2 += v * v;
  }
#pragma unroll
  for (int m = 1; m <= 16; m <<= 1) { s += __shfl_xor(s, m); s2 += __shfl_xor(s2, m); }
  float mean = s * (1.0f / 512.0f);
  float var = s2 * (1.0f / 512.0f) - mean * mean;
  float rstd = rsqrtf(var + 1e-5f);
#pragma unroll
  for (int i = 0; i < 16; ++i) {
    int c = lane + 32 * i;
    float t = (vals[i] - mean) * rstd * g[c] + b[c];
    float ge = 0.5f * t * (1.0f + erff(t * 0.70710678118f));
    y[(size_t)row * 512 + c] = (bf16)ge;
  }
}

// conf[r] = b2 + sum_d relu(h[r,d]) * w2[d]   (wave per row)
__global__ __launch_bounds__(128)
void conf_kernel(const float* __restrict__ h, const float* __restrict__ w2,
                 const float* __restrict__ b2, float* __restrict__ out, int R) {
  int wave = threadIdx.x >> 5, lane = threadIdx.x & 31;
  int row = blockIdx.x * 4 + wave;
  if (row >= R) return;
  const float* hr = h + (size_t)row * D_;
  float acc = 0.0f;
#pragma unroll
  for (int i = 0; i < 8; ++i) {
    int c = lane + 32 * i;
    float v = fmaxf(hr[c], 0.0f);
    acc += v * w2[c];
  }
#pragma unroll
  for (int m = 1; m <= 16; m <<= 1) acc += __shfl_xor(acc, m);
  if (lane == 0) out[row] = acc + b2[0];
}

// =====================================================================
extern "C" void kernel_launch(void* const* d_in, const int* in_sizes, int n_in,
                              void* d_out, int out_size, void* d_ws, size_t ws_size,
                              hipStream_t stream) {
  (void)in_sizes; (void)n_in; (void)out_size; (void)ws_size;
  typedef long long ll;

  // ---- inputs ----
  const float* t_desc = (const float*)d_in[0];
  const float* s_desc = (const float*)d_in[1];
  const float* t_enc  = (const float*)d_in[2];
  const float* s_enc  = (const float*)d_in[3];
  const float* q_enc  = (const float*)d_in[4];
  const float* k_enc  = (const float*)d_in[5];

  struct Blk {
    const float *qkv_w, *qkv_b, *out_w, *out_b, *f1_w, *f1_b, *ln_g, *ln_b, *f2_w, *f2_b;
  };
  auto getBlk = [&](int base) {
    Blk w;
    w.qkv_w = (const float*)d_in[base + 0]; w.qkv_b = (const float*)d_in[base + 1];
    w.out_w = (const float*)d_in[base + 2]; w.out_b = (const float*)d_in[base + 3];
    w.f1_w  = (const float*)d_in[base + 4]; w.f1_b  = (const float*)d_in[base + 5];
    w.ln_g  = (const float*)d_in[base + 6]; w.ln_b  = (const float*)d_in[base + 7];
    w.f2_w  = (const float*)d_in[base + 8]; w.f2_b  = (const float*)d_in[base + 9];
    return w;
  };
  Blk sa = getBlk(6), ssb = getBlk(16), tt = getBlk(26);
  const float* ca_q_w  = (const float*)d_in[36]; const float* ca_q_b  = (const float*)d_in[37];
  const float* ca_kv_w = (const float*)d_in[38]; const float* ca_kv_b = (const float*)d_in[39];
  const float* ca_out_w = (const float*)d_in[40]; const float* ca_out_b = (const float*)d_in[41];
  const float* ca_f1_w = (const float*)d_in[42]; const float* ca_f1_b = (const float*)d_in[43];
  const float* ca_ln_g = (const float*)d_in[44]; const float* ca_ln_b = (const float*)d_in[45];
  const float* ca_f2_w = (const float*)d_in[46]; const float* ca_f2_b = (const float*)d_in[47];
  const float* cf_w1 = (const float*)d_in[48]; const float* cf_b1 = (const float*)d_in[49];
  const float* cf_w2 = (const float*)d_in[50]; const float* cf_b2 = (const float*)d_in[51];

  // ---- workspace bump allocator ----
  size_t off = 0;
  auto alloc = [&](size_t bytes) -> void* {
    off = (off + 255) & ~(size_t)255;
    void* p = (char*)d_ws + off;
    off += bytes;
    return p;
  };
  const ll RD  = (ll)R_ * D_;        // 6144*256
  const ll R2D = (ll)R_ * 2 * D_;
  const ll R3D = (ll)R_ * 3 * D_;
  const ll S2SQ = (ll)B_ * H_ * NV_ * NV_ * NS_ * HD_;   // 9437184
  const ll T2TVT = (ll)B_ * M_ * H_ * HD_ * 32;           // 8388608

  // bf16 transposed weights Wt[N][K]
  struct BW { bf16 *qkvW, *outW, *f1W, *f2W; };
  auto allocBW = [&]() {
    BW w;
    w.qkvW = (bf16*)alloc((size_t)D_ * 3 * D_ * 2);
    w.outW = (bf16*)alloc((size_t)D_ * D_ * 2);
    w.f1W  = (bf16*)alloc((size_t)2 * D_ * 2 * D_ * 2);
    w.f2W  = (bf16*)alloc((size_t)2 * D_ * D_ * 2);
    return w;
  };
  BW saB = allocBW(), ssB = allocBW(), ttB = allocBW();
  bf16* caQw  = (bf16*)alloc((size_t)D_ * D_ * 2);
  bf16* caKVw = (bf16*)alloc((size_t)D_ * 2 * D_ * 2);
  bf16* caOw  = (bf16*)alloc((size_t)D_ * D_ * 2);
  bf16* caF1w = (bf16*)alloc((size_t)2 * D_ * 2 * D_ * 2);
  bf16* caF2w = (bf16*)alloc((size_t)2 * D_ * D_ * 2);
  bf16* cfW1b = (bf16*)alloc((size_t)D_ * D_ * 2);

  // activations
  bf16* xb    = (bf16*)alloc(RD * 2);
  bf16* tb    = (bf16*)alloc(RD * 2);
  bf16* sb    = (bf16*)alloc(RD * 2);
  bf16* catb  = (bf16*)alloc(R2D * 2);
  bf16* h2b   = (bf16*)alloc(R2D * 2);
  bf16* ctxb  = (bf16*)alloc(RD * 2);
  bf16* qb    = (bf16*)alloc(S2SQ * 2);
  bf16* kb    = (bf16*)alloc(RD * 2);
  bf16* vtb   = (bf16*)alloc(T2TVT * 2);
  bf16* s2sct = (bf16*)alloc(S2SQ * 2);
  float* qkvF = (float*)alloc(R3D * 4);
  float* msgF = (float*)alloc(RD * 4);
  float* h1F  = (float*)alloc(R2D * 4);
  float* f2F  = (float*)alloc(RD * 4);
  float* t1   = (float*)alloc(RD * 4);
  float* s1   = (float*)alloc(RD * 4);
  float* tA   = (float*)alloc(RD * 4);
  float* sA   = (float*)alloc(RD * 4);

  float* outT2 = (float*)d_out;
  float* outS2 = outT2 + RD;
  float* outCf = outT2 + 2 * RD;

  // ---- host helpers ----
  auto cvt = [&](const float* src, bf16* dst, ll n) {
    cvt_f32_bf16_kernel<<<dim3((unsigned)((n + 255) / 256)), dim3(256), 0, stream>>>(src, dst, n);
  };
  auto cvtT = [&](const float* src, bf16* dst, int Kd, int Nd) {
    ll n = (ll)Kd * Nd;
    cvt_transpose_bf16_kernel<<<dim3((unsigned)((n + 255) / 256)), dim3(256), 0, stream>>>(src, dst, Kd, Nd);
  };
  auto fillz = [&](bf16* dst, ll n) {
    fill_bf16_kernel<<<dim3((unsigned)((n + 255) / 256)), dim3(256), 0, stream>>>(dst, n);
  };
  auto gemm = [&](const bf16* A, const bf16* Wt, const float* bias, float* C, int M, int N, int K) {
    gemm_bf16_kernel<<<dim3((unsigned)(N / 64), (unsigned)(M / 128)), dim3(256), 0, stream>>>(A, Wt, bias, C, M, N, K);
  };
  auto attn = [&](const bf16* q, const bf16* k, const bf16* vt, bf16* out,
                  int nBatch, int Lq, int Lk, int LkPad, float scale, int nMid, int nInner,
                  ll S0, ll S1, ll S2, ll SR) {
    ll tasks = (ll)nBatch * ((Lq + 15) / 16);
    attn_kernel<<<dim3((unsigned)((tasks + 3) / 4)), dim3(128), 0, stream>>>(
        q, k, vt, out, nBatch, Lq, Lk, LkPad, scale, nMid, nInner, S0, S1, S2, SR);
  };
  auto tail = [&](const float* x, const bf16* outW, const float* outBias,
                  const bf16* f1W, const float* f1B, const float* lnG, const float* lnB,
                  const bf16* f2W, const float* f2B, float* dst) {
    gemm(ctxb, outW, outBias, msgF, R_, D_, D_);
    concat_bf16_kernel<<<dim3((unsigned)(R2D / 256)), dim3(256), 0, stream>>>(x, msgF, catb, R2D);
    gemm(catb, f1W, f1B, h1F, R_, 2 * D_, 2 * D_);
    ln_gelu_kernel<<<dim3(R_ / 4), dim3(128), 0, stream>>>(h1F, lnG, lnB, h2b, R_);
    gemm(h2b, f2W, f2B, f2F, R_, D_, 2 * D_);
    add_res_kernel<<<dim3((unsigned)(RD / 256)), dim3(256), 0, stream>>>(x, f2F, dst, RD);
  };

  // ---- convert + transpose weights once (Wt[N][K]) ----
  auto cvtBlk = [&](const Blk& w, const BW& b) {
    cvtT(w.qkv_w, b.qkvW, D_, 3 * D_);
    cvtT(w.out_w, b.outW, D_, D_);
    cvtT(w.f1_w, b.f1W, 2 * D_, 2 * D_);
    cvtT(w.f2_w, b.f2W, 2 * D_, D_);
  };
  cvtBlk(sa, saB); cvtBlk(ssb, ssB); cvtBlk(tt, ttB);
  cvtT(ca_q_w, caQw, D_, D_);
  cvtT(ca_kv_w, caKVw, D_, 2 * D_);
  cvtT(ca_out_w, caOw, D_, D_);
  cvtT(ca_f1_w, caF1w, 2 * D_, 2 * D_);
  cvtT(ca_f2_w, caF2w, 2 * D_, D_);
  cvtT(cf_w1, cfW1b, D_, D_);

  const ll selfS0 = (ll)NV_ * M_ * D_, selfS1 = (ll)M_ * D_;

  // ---- self blocks (shared sa weights) ----
  auto selfBlock = [&](const float* x, const float* enc, float* dst) {
    cvt(x, xb, RD);
    gemm(xb, saB.qkvW, sa.qkv_b, qkvF, R_, 3 * D_, D_);
    ll gt = (ll)B_ * NV_ * H_ * M_ * 32;
    gather_rope_self_kernel<<<dim3((unsigned)(gt / 256)), dim3(256), 0, stream>>>(qkvF, enc, qb, kb, vtb, M_);
    attn(qb, kb, vtb, ctxb, B_ * NV_ * H_, M_, M_, M_, 0.125f, NV_, H_, selfS0, selfS1, HD_, D_);
    tail(x, saB.outW, sa.out_b, saB.f1W, sa.f1_b, sa.ln_g, sa.ln_b, saB.f2W, sa.f2_b, dst);
  };
  selfBlock(t_desc, t_enc, t1);
  selfBlock(s_desc, s_enc, s1);

  // ---- t2t block (attention over the 6 views per keypoint) ----
  cvt(t1, xb, RD);
  gemm(xb, ttB.qkvW, tt.qkv_b, qkvF, R_, 3 * D_, D_);
  fillz(vtb, T2TVT);
  {
    ll gt = (ll)B_ * M_ * H_ * NV_ * 32;
    gather_t2t_kernel<<<dim3((unsigned)(gt / 256)), dim3(256), 0, stream>>>(qkvF, qb, kb, vtb);
  }
  attn(qb, kb, vtb, ctxb, B_ * M_ * H_, NV_, NV_, 32, 0.125f, M_, H_,
       (ll)NV_ * M_ * D_, (ll)D_, (ll)HD_, (ll)M_ * D_);
  tail(t1, ttB.outW, tt.out_b, ttB.f1W, tt.f1_b, tt.ln_g, tt.ln_b, ttB.f2W, tt.f2_b, tA);

  // ---- s2s block ----
  cvt(s1, xb, RD);
  gemm(xb, ssB.qkvW, ssb.qkv_b, qkvF, R_, 3 * D_, D_);
  {
    ll gt = (ll)B_ * H_ * NV_ * NS_ * 32;
    gather_s2s_kv_kernel<<<dim3((unsigned)(gt / 256)), dim3(256), 0, stream>>>(qkvF, k_enc, kb, vtb);
    ll gq = (ll)B_ * H_ * NV_ * NV_ * NS_ * 32;
    gather_s2s_q_kernel<<<dim3((unsigned)(gq / 256)), dim3(256), 0, stream>>>(qkvF, q_enc, qb);
  }
  attn(qb, kb, vtb, s2sct, B_ * H_ * NV_, NV_ * NS_, NS_, NS_, 0.125f, H_, NV_,
       (ll)H_ * NV_ * NV_ * NS_ * HD_, (ll)NV_ * NV_ * NS_ * HD_, (ll)NV_ * NS_ * HD_, (ll)HD_);
  s2s_aggregate_kernel<<<dim3((unsigned)(RD / 256)), dim3(256), 0, stream>>>(s2sct, ctxb);
  tail(s1, ssB.outW, ssb.out_b, ssB.f1W, ssb.f1_b, ssb.ln_g, ssb.ln_b, ssB.f2W, ssb.f2_b, sA);

  // ---- cross blocks (shared ca weights) ----
  cvt(tA, tb, RD);
  cvt(sA, sb, RD);
  float* qF = qkvF;          // [R, 256]
  float* kvF = qkvF + RD;    // [R, 512]
  auto crossBlock = [&](const float* x0, const bf16* x0b, const bf16* x1b, float* dst) {
    gemm(x0b, caQw, ca_q_b, qF, R_, D_, D_);
    gemm(x1b, caKVw, ca_kv_b, kvF, R_, 2 * D_, D_);
    ll gt = (ll)B_ * NV_ * H_ * M_ * 32;
    gather_cross_kernel<<<dim3((unsigned)(gt / 256)), dim3(256), 0, stream>>>(qF, kvF, qb, kb, vtb);
    attn(qb, kb, vtb, ctxb, B_ * NV_ * H_, M_, M_, M_, 0.125f, NV_, H_, selfS0, selfS1, HD_, D_);
    tail(x0, caOw, ca_out_b, caF1w, ca_f1_b, ca_ln_g, ca_ln_b, caF2w, ca_f2_b, dst);
  };
  crossBlock(tA, tb, sb, outT2);
  crossBlock(sA, sb, tb, outS2);

  // ---- confidence head: linear(relu(linear(sA))) ----
  gemm(sb, cfW1b, cf_b1, msgF, R_, D_, D_);
  conf_kernel<<<dim3(R_ / 4), dim3(128), 0, stream>>>(msgF, cf_w2, cf_b2, outCf, R_);
}
